// DGCNN_Grouper_87617332838602
// MI455X (gfx1250) — matrix operand order repaired and checked by reading, not compiled, not run on hardware
//
#include <hip/hip_runtime.h>

typedef __attribute__((ext_vector_type(16))) _Float16 v16h;
typedef __attribute__((ext_vector_type(8)))  float    v8f;
typedef __attribute__((ext_vector_type(2)))  float    v2f;

#define GN_EPS 1e-5f

// ---------------------------------------------------------------------------
// Stage A: input_trans Conv1d 3->8 (+bias)
// ---------------------------------------------------------------------------
__global__ void k_input_trans(const float* __restrict__ x,
                              const float* __restrict__ w,
                              const float* __restrict__ bias,
                              float* __restrict__ f0, int B, int N) {
  int i = blockIdx.x * blockDim.x + threadIdx.x;
  if (i >= B * N) return;
  int b = i / N, n = i % N;
  float c0 = x[((size_t)b * 3 + 0) * N + n];
  float c1 = x[((size_t)b * 3 + 1) * N + n];
  float c2 = x[((size_t)b * 3 + 2) * N + n];
#pragma unroll
  for (int o = 0; o < 8; ++o) {
    f0[((size_t)b * 8 + o) * N + n] =
        w[o * 3 + 0] * c0 + w[o * 3 + 1] * c1 + w[o * 3 + 2] * c2 + bias[o];
  }
}

// ---------------------------------------------------------------------------
// Pack stage weights W[O,2C] (f32) into f16 A-matrix fragments laid out in the
// contiguous k order k = 16*(lane/16) + h, same as the LDS B operand, zero-padded to K%32==0.
// Layout: wf[(mt*nks+ks)*512 + lane*16 + h]  -> one v16h per (frag,lane).
// ---------------------------------------------------------------------------
__global__ void k_packw(const float* __restrict__ W, _Float16* __restrict__ wf,
                        int O, int K2, int Kpad) {
  int nks = Kpad / 32;
  int total = (O / 16) * nks * 512;
  int i = blockIdx.x * blockDim.x + threadIdx.x;
  if (i >= total) return;
  int h = i & 15;
  int lane = (i >> 4) & 31;
  int frag = i >> 9;
  int ks = frag % nks, mt = frag / nks;
  int half = lane >> 4;
  int v = h >> 1, p = h & 1;
  int ka = half * 16 + 2 * v + p;
  int kk = ks * 32 + ka;
  int orow = mt * 16 + (lane & 15);
  float wv = (kk < K2) ? W[(size_t)orow * K2 + kk] : 0.0f;
  wf[i] = (_Float16)wv;
}

// ---------------------------------------------------------------------------
// KNN: one wave32 per 16-query tile. Gram tile via V_WMMA_F32_16X16X4_F32.
// Top-16 kept per owner lane in REGISTERS as packed u64 keys
// [monotonic(dist):32 | index:32]; branchless min/max bubble insert with a
// cheap reject early-out. idx layout [B,16,Nq].
// ---------------------------------------------------------------------------
__global__ __launch_bounds__(32) void k_knn(const float* __restrict__ cq,
                                            const float* __restrict__ ck,
                                            int* __restrict__ idx_out,
                                            int Nq, int Nk) {
  int b = blockIdx.y;
  int q0 = blockIdx.x * 16;
  int lane = threadIdx.x;
  int m = lane & 15;
  int half = lane >> 4;

  __shared__ float qq[16];
  __shared__ float rr[16];
  __shared__ float qst[3][16];
  __shared__ float kst[3][16];
  __shared__ __align__(16) float dq[16][20];  // [query][key], 16B-aligned rows

  const float* cqb = cq + (size_t)b * 3 * Nq;
  const float* ckb = ck + (size_t)b * 3 * Nk;

  float qx = 0.f, qy = 0.f;
  if (lane < 16) {
    qx = cqb[0 * Nq + q0 + m];
    qy = cqb[1 * Nq + q0 + m];
    float qz = cqb[2 * Nq + q0 + m];
    qst[0][m] = qx; qst[1][m] = qy; qst[2][m] = qz;
    qq[m] = qx * qx + qy * qy + qz * qz;
  }
  unsigned long long lst[16];
#pragma unroll
  for (int j = 0; j < 16; ++j) lst[j] = ~0ull;
  __syncthreads();

  // A-matrix (f32 16x4): lanes 0-15 hold K=0,1 ; lanes 16-31 hold K=2,3
  v2f a;
  if (half == 0) { a.x = qx; a.y = qy; }
  else           { a.x = qst[2][m]; a.y = 0.0f; }

  int ntiles = Nk / 16;
  for (int t = 0; t < ntiles; ++t) {
    int k0 = t * 16;
    if (lane < 16) {
      float rx = ckb[0 * Nk + k0 + m];
      float ry = ckb[1 * Nk + k0 + m];
      float rz = ckb[2 * Nk + k0 + m];
      kst[0][m] = rx; kst[1][m] = ry; kst[2][m] = rz;
      rr[m] = rx * rx + ry * ry + rz * rz;
    }
    if (t + 1 < ntiles) __builtin_prefetch(&ckb[0 * Nk + k0 + 16], 0, 1);
    __syncthreads();

    v8f c = {};
#if __has_builtin(__builtin_amdgcn_wmma_f32_16x16x4_f32)
    // B-matrix (f32 4x16): vgpr v, K = v + 2*half, N = lane%16
    v2f bm;
    int kb = half << 1;
    bm.x = (kb < 3)     ? kst[kb][m]     : 0.0f;
    bm.y = (kb + 1 < 3) ? kst[kb + 1][m] : 0.0f;
    c = __builtin_amdgcn_wmma_f32_16x16x4_f32(false, a, false, bm, (short)0,
                                              c, false, false);
#else
#pragma unroll
    for (int g = 0; g < 8; ++g) {
      int M = half * 8 + g;
      c[g] = qst[0][M] * kst[0][m] + qst[1][M] * kst[1][m] + qst[2][M] * kst[2][m];
    }
#endif
    // C/D layout: vgpr g -> M = g + 8*half, N = lane%16
#pragma unroll
    for (int g = 0; g < 8; ++g) {
      int M = half * 8 + g;
      dq[M][m] = qq[M] + rr[m] - 2.0f * c[g];
    }
    __syncthreads();

    if (lane < 16) {
      const float4* row = (const float4*)&dq[m][0];
      for (int nn = 0; nn < 4; ++nn) {  // rolled: bounds static footprint
        float4 r4 = row[nn];
        float dv0 = r4.x, dv1 = r4.y, dv2 = r4.z, dv3 = r4.w;
#pragma unroll
        for (int u = 0; u < 4; ++u) {
          float d = (u == 0) ? dv0 : (u == 1) ? dv1 : (u == 2) ? dv2 : dv3;
          unsigned kb2 = __float_as_uint(d);
          kb2 = (kb2 & 0x80000000u) ? ~kb2 : (kb2 | 0x80000000u);
          unsigned long long key =
              ((unsigned long long)kb2 << 32) | (unsigned)(k0 + nn * 4 + u);
          if (key < lst[15]) {
#pragma unroll
            for (int j = 0; j < 16; ++j) {
              bool sw = key < lst[j];
              unsigned long long lo = sw ? key : lst[j];
              unsigned long long hi = sw ? lst[j] : key;
              lst[j] = lo;
              key = hi;
            }
          }
        }
      }
    }
    __syncthreads();
  }

  if (lane < 16) {
#pragma unroll
    for (int j = 0; j < 16; ++j)
      idx_out[((size_t)b * 16 + j) * Nq + q0 + m] =
          (int)(unsigned)(lst[j] & 0xFFFFFFFFu);
  }
}

// ---------------------------------------------------------------------------
// Edge conv: one wave32 per (b,q). Builds h=[fk[nb]-fq ; fq] (2C x 16)
// column-major as f16 in LDS (32B-aligned rows), then per k-step:
//   A operand: one 32B contiguous global load from pre-packed wf
//   B operand: one 32B contiguous LDS load (lane's column, K-slice)
//   -> V_WMMA_F32_16X16X32_F16
// Output conv[B,O,Nq,16] f32. Neighbor gather reproduces the reference's
// flat-reshape scramble: column (q,j) uses idx[b, (q*16+j)/Nq, (q*16+j)%Nq].
// ---------------------------------------------------------------------------
__global__ __launch_bounds__(32) void k_edgeconv(
    const float* __restrict__ fq, const float* __restrict__ fk,
    const int* __restrict__ idx, const _Float16* __restrict__ wf,
    float* __restrict__ out, int C, int O, int Nq, int Nk) {
  int q = blockIdx.x;
  int b = blockIdx.y;
  int lane = threadIdx.x;
  int half = lane >> 4;
  int n = lane & 15;
  int K2 = 2 * C;
  int Kpad = (K2 + 31) & ~31;
  int nks = Kpad / 32;

  // column-major h: hk2[col][K], row stride 144 halves = 288B (32B aligned)
  __shared__ __align__(32) _Float16 hk2[16][144];
  __shared__ float fqv[64];  // C <= 64
  __shared__ int nb[16];

  const float* fqb = fq + (size_t)b * C * Nq;
  const float* fkb = fk + (size_t)b * C * Nk;

  if (lane < 16) {
    int lin = q * 16 + lane;              // flat position (q,j)
    int r = lin / Nq, cc = lin % Nq;      // reference's reshape quirk
    nb[lane] = idx[((size_t)b * 16 + r) * Nq + cc];
  }
  for (int c = lane; c < C; c += 32) fqv[c] = fqb[(size_t)c * Nq + q];
  for (int e = lane; e < (Kpad - K2) * 16; e += 32)
    hk2[e & 15][K2 + (e >> 4)] = (_Float16)0.0f;
  __syncthreads();

  for (int e = lane; e < C * 16; e += 32) {
    int c = e >> 4, j = e & 15;
    float fv = fqv[c];
    float nv = fkb[(size_t)c * Nk + nb[j]];
    hk2[j][c]     = (_Float16)(nv - fv);
    hk2[j][C + c] = (_Float16)fv;
  }
  __syncthreads();

  int nmt = O / 16;
  const _Float16* wl = wf + (size_t)lane * 16;
  for (int mt = 0; mt < nmt; ++mt) {
    v8f acc = {};
    for (int ks = 0; ks < nks; ++ks) {
      v16h av = *(const v16h*)(wl + (size_t)(mt * nks + ks) * 512);
      v16h bv = *(const v16h*)&hk2[n][ks * 32 + half * 16];
      acc = __builtin_amdgcn_wmma_f32_16x16x32_f16(false, av, false, bv,
                                                   (short)0, acc, false, false);
    }
#pragma unroll
    for (int g = 0; g < 8; ++g) {
      int o = mt * 16 + half * 8 + g;
      out[(((size_t)b * O + o) * Nq + q) * 16 + n] = acc[g];
    }
  }
}

// ---------------------------------------------------------------------------
// GroupNorm stats: one block per (group, b); group channels are contiguous.
// ---------------------------------------------------------------------------
__global__ __launch_bounds__(256) void k_gnstats(const float* __restrict__ conv,
                                                 float* __restrict__ stats,
                                                 int O, int Nq) {
  int g = blockIdx.x, b = blockIdx.y;
  int Cg = O / 4;
  size_t base = (((size_t)b * O + (size_t)g * Cg) * Nq) * 16;
  size_t total = (size_t)Cg * Nq * 16;
  float s = 0.f, s2 = 0.f;
  for (size_t i = threadIdx.x; i < total; i += 256) {
    float v = conv[base + i];
    s += v; s2 += v * v;
  }
  __shared__ float rs[256], rs2[256];
  rs[threadIdx.x] = s; rs2[threadIdx.x] = s2;
  __syncthreads();
  for (int st = 128; st > 0; st >>= 1) {
    if ((int)threadIdx.x < st) {
      rs[threadIdx.x] += rs[threadIdx.x + st];
      rs2[threadIdx.x] += rs2[threadIdx.x + st];
    }
    __syncthreads();
  }
  if (threadIdx.x == 0) {
    float mean = rs[0] / (float)total;
    float var = rs2[0] / (float)total - mean * mean;
    stats[(b * 4 + g) * 2 + 0] = mean;
    stats[(b * 4 + g) * 2 + 1] = var;
  }
}

// ---------------------------------------------------------------------------
// GroupNorm apply + LeakyReLU(0.2) + max over k=16
// ---------------------------------------------------------------------------
__global__ void k_gnmax(const float* __restrict__ conv,
                        const float* __restrict__ stats,
                        const float* __restrict__ gamma,
                        const float* __restrict__ beta,
                        float* __restrict__ fout, int B, int O, int Nq) {
  int i = blockIdx.x * blockDim.x + threadIdx.x;
  if (i >= B * O * Nq) return;
  int b = i / (O * Nq);
  int o = (i / Nq) % O;
  int g = o / (O / 4);
  float mean = stats[(b * 4 + g) * 2 + 0];
  float var = stats[(b * 4 + g) * 2 + 1];
  float scale = gamma[o] * rsqrtf(var + GN_EPS);
  float shift = beta[o] - mean * scale;
  float mx = -3.0e38f;
#pragma unroll
  for (int j = 0; j < 16; ++j) {
    float v = conv[(size_t)i * 16 + j] * scale + shift;
    v = (v > 0.f) ? v : 0.2f * v;
    mx = fmaxf(mx, v);
  }
  fout[i] = mx;
}

// ---------------------------------------------------------------------------
// Furthest point sampling (pointnet2 semantics, start at 0). One block per b.
// ---------------------------------------------------------------------------
__global__ __launch_bounds__(256) void k_fps(const float* __restrict__ coor,
                                             int* __restrict__ fidx,
                                             int N, int m) {
  int b = blockIdx.x;
  int tid = threadIdx.x;
  __shared__ float dist[2048];
  __shared__ float rv[256];
  __shared__ int ri[256];
  __shared__ int sfar;
  const float* cb = coor + (size_t)b * 3 * N;
  for (int n = tid; n < N; n += 256) dist[n] = 1e10f;
  if (tid == 0) sfar = 0;
  __syncthreads();
  for (int i = 0; i < m; ++i) {
    int far = sfar;
    if (tid == 0) fidx[b * m + i] = far;
    float cx = cb[0 * N + far], cy = cb[1 * N + far], cz = cb[2 * N + far];
    float bestv = -1.f; int besti = 0x7fffffff;
    for (int n = tid; n < N; n += 256) {
      float dx = cb[0 * N + n] - cx;
      float dy = cb[1 * N + n] - cy;
      float dz = cb[2 * N + n] - cz;
      float d = dx * dx + dy * dy + dz * dz;
      float nd = fminf(dist[n], d);
      dist[n] = nd;
      if (nd > bestv) { bestv = nd; besti = n; }
    }
    rv[tid] = bestv; ri[tid] = besti;
    __syncthreads();
    for (int st = 128; st > 0; st >>= 1) {
      if (tid < st) {
        if (rv[tid + st] > rv[tid] ||
            (rv[tid + st] == rv[tid] && ri[tid + st] < ri[tid])) {
          rv[tid] = rv[tid + st]; ri[tid] = ri[tid + st];
        }
      }
      __syncthreads();
    }
    if (tid == 0) sfar = ri[0];
    __syncthreads();
  }
}

// ---------------------------------------------------------------------------
// fps_downsample gather: comb=[coor;f][:, idx]
// ---------------------------------------------------------------------------
__global__ void k_gather(const float* __restrict__ coor,
                         const float* __restrict__ f,
                         const int* __restrict__ fidx,
                         float* __restrict__ coor_out,
                         float* __restrict__ f_out,
                         int B, int C, int N, int m) {
  int i = blockIdx.x * blockDim.x + threadIdx.x;
  int tot = (3 + C) * m;
  if (i >= B * tot) return;
  int b = i / tot;
  int r = i % tot;
  int ch = r / m, mi = r % m;
  int src = fidx[b * m + mi];
  if (ch < 3)
    coor_out[((size_t)b * 3 + ch) * m + mi] = coor[((size_t)b * 3 + ch) * N + src];
  else
    f_out[((size_t)b * C + (ch - 3)) * m + mi] = f[((size_t)b * C + (ch - 3)) * N + src];
}

// ---------------------------------------------------------------------------
extern "C" void kernel_launch(void* const* d_in, const int* in_sizes, int n_in,
                              void* d_out, int out_size, void* d_ws, size_t ws_size,
                              hipStream_t stream) {
  (void)in_sizes; (void)n_in; (void)out_size; (void)ws_size;
  const float* x    = (const float*)d_in[0];
  const float* w_in = (const float*)d_in[1];
  const float* b_in = (const float*)d_in[2];
  const float* w1 = (const float*)d_in[3];
  const float* g1 = (const float*)d_in[4];
  const float* be1 = (const float*)d_in[5];
  const float* w2 = (const float*)d_in[6];
  const float* g2 = (const float*)d_in[7];
  const float* be2 = (const float*)d_in[8];
  const float* w3 = (const float*)d_in[9];
  const float* g3 = (const float*)d_in[10];
  const float* be3 = (const float*)d_in[11];
  const float* w4 = (const float*)d_in[12];
  const float* g4 = (const float*)d_in[13];
  const float* be4 = (const float*)d_in[14];

  const int B = 8, N = 2048;
  const int N4 = N / 4, N16 = N / 16;

  char* ws = (char*)d_ws;
  size_t off = 0;
  auto alloc = [&](size_t bytes) -> void* {
    void* p = ws + off;
    off += (bytes + 255) & ~(size_t)255;
    return p;
  };
  float* f0   = (float*)alloc((size_t)B * 8 * N * 4);
  float* f1   = (float*)alloc((size_t)B * 32 * N * 4);
  float* fq1  = (float*)alloc((size_t)B * 32 * N4 * 4);
  float* cq1  = (float*)alloc((size_t)B * 3 * N4 * 4);
  float* f2   = (float*)alloc((size_t)B * 64 * N4 * 4);
  float* f3   = (float*)alloc((size_t)B * 64 * N4 * 4);
  float* fq2  = (float*)alloc((size_t)B * 64 * N16 * 4);
  int* idxb   = (int*)alloc((size_t)B * 16 * N * 4);
  int* fpsi   = (int*)alloc((size_t)B * N4 * 4);
  float* stats = (float*)alloc((size_t)B * 4 * 2 * 4);
  // packed f16 A-fragments per stage: (O/16)*(Kpad/32)*512 halves
  const int t1 = (32 / 16) * (32 / 32) * 512;
  const int t2 = (64 / 16) * (64 / 32) * 512;
  const int t3 = (64 / 16) * (128 / 32) * 512;
  const int t4 = (128 / 16) * (128 / 32) * 512;
  _Float16* wf1 = (_Float16*)alloc((size_t)t1 * 2);
  _Float16* wf2 = (_Float16*)alloc((size_t)t2 * 2);
  _Float16* wf3 = (_Float16*)alloc((size_t)t3 * 2);
  _Float16* wf4 = (_Float16*)alloc((size_t)t4 * 2);
  float* conv = (float*)alloc((size_t)B * 32 * N * 16 * 4);  // largest reuse

  float* out_coor = (float*)d_out;                     // [8,3,128]
  float* out_f = (float*)d_out + (size_t)B * 3 * N16;  // [8,128,128]

  // Pack all stage weights up front (constants, independent of data flow)
  k_packw<<<(t1 + 255) / 256, 256, 0, stream>>>(w1, wf1, 32, 16, 32);
  k_packw<<<(t2 + 255) / 256, 256, 0, stream>>>(w2, wf2, 64, 64, 64);
  k_packw<<<(t3 + 255) / 256, 256, 0, stream>>>(w3, wf3, 64, 128, 128);
  k_packw<<<(t4 + 255) / 256, 256, 0, stream>>>(w4, wf4, 128, 128, 128);

  // Stage A: input_trans
  k_input_trans<<<(B * N + 255) / 256, 256, 0, stream>>>(x, w_in, b_in, f0, B, N);

  // Stage B: edge layer 1 (C=8 -> O=32) at N=2048
  k_knn<<<dim3(N / 16, B), 32, 0, stream>>>(x, x, idxb, N, N);
  k_edgeconv<<<dim3(N, B), 32, 0, stream>>>(f0, f0, idxb, wf1, conv, 8, 32, N, N);
  k_gnstats<<<dim3(4, B), 256, 0, stream>>>(conv, stats, 32, N);
  k_gnmax<<<(B * 32 * N + 255) / 256, 256, 0, stream>>>(conv, stats, g1, be1, f1, B, 32, N);

  // fps to N/4
  k_fps<<<B, 256, 0, stream>>>(x, fpsi, N, N4);
  k_gather<<<(B * (3 + 32) * N4 + 255) / 256, 256, 0, stream>>>(x, f1, fpsi, cq1, fq1, B, 32, N, N4);

  // Stage C: edge layer 2 (C=32 -> O=64), queries 512 vs keys 2048
  k_knn<<<dim3(N4 / 16, B), 32, 0, stream>>>(cq1, x, idxb, N4, N);
  k_edgeconv<<<dim3(N4, B), 32, 0, stream>>>(fq1, f1, idxb, wf2, conv, 32, 64, N4, N);
  k_gnstats<<<dim3(4, B), 256, 0, stream>>>(conv, stats, 64, N4);
  k_gnmax<<<(B * 64 * N4 + 255) / 256, 256, 0, stream>>>(conv, stats, g2, be2, f2, B, 64, N4);

  // Stage D: edge layer 3 (C=64 -> O=64), self at 512
  k_knn<<<dim3(N4 / 16, B), 32, 0, stream>>>(cq1, cq1, idxb, N4, N4);
  k_edgeconv<<<dim3(N4, B), 32, 0, stream>>>(f2, f2, idxb, wf3, conv, 64, 64, N4, N4);
  k_gnstats<<<dim3(4, B), 256, 0, stream>>>(conv, stats, 64, N4);
  k_gnmax<<<(B * 64 * N4 + 255) / 256, 256, 0, stream>>>(conv, stats, g3, be3, f3, B, 64, N4);

  // fps to N/16 (coor part goes straight into d_out)
  k_fps<<<B, 256, 0, stream>>>(cq1, fpsi, N4, N16);
  k_gather<<<(B * (3 + 64) * N16 + 255) / 256, 256, 0, stream>>>(cq1, f3, fpsi, out_coor, fq2, B, 64, N4, N16);

  // Stage E: edge layer 4 (C=64 -> O=128), queries 128 vs keys 512
  k_knn<<<dim3(N16 / 16, B), 32, 0, stream>>>(out_coor, cq1, idxb, N16, N4);
  k_edgeconv<<<dim3(N16, B), 32, 0, stream>>>(fq2, f3, idxb, wf4, conv, 64, 128, N16, N4);
  k_gnstats<<<dim3(4, B), 256, 0, stream>>>(conv, stats, 128, N16);
  k_gnmax<<<(B * 128 * N16 + 255) / 256, 256, 0, stream>>>(conv, stats, g4, be4, out_f, B, 128, N16);
}
